// Reformer_64321430225699
// MI455X (gfx1250) — matrix operand order repaired and checked by reading, not compiled, not run on hardware
//
#include <hip/hip_runtime.h>
#include <hip/hip_bf16.h>
#include <math.h>

typedef _Float16 half_t;
typedef __attribute__((ext_vector_type(16))) _Float16 v16h;
typedef __attribute__((ext_vector_type(8)))  float    v8f;

#define BATCH   8
#define SEQ     4096
#define DMODEL  256
#define NHEAD   8
#define DH      32
#define NLAYERS 4
#define NCLASS  10
#define BUCKSZ  64
#define NHASH   2
#define NBUCK   64                 /* SEQ / BUCKSZ */
#define NCH     (NHASH * NBUCK)    /* 128 chunks per (b,h) */
#define SEQ2    (NHASH * SEQ)      /* 8192 */
#define MROWS   (BATCH * SEQ)      /* 32768 */
#define DFF     (4 * DMODEL)       /* 1024 */
#define KC      128                /* GEMM K-chunk staged in LDS */

#define FLAG_GELU  1
#define FLAG_ACCUM 2

__device__ __forceinline__ float gelu_f(float x) {
    float x3 = x * x * x;
    return 0.5f * x * (1.0f + tanhf(0.7978845608028654f * (x + 0.044715f * x3)));
}

// ---------------------------------------------------------------- embedding
__global__ void k_embed(const int* __restrict__ x, const float* __restrict__ emb,
                        const float* __restrict__ pos, float* __restrict__ x1,
                        float* __restrict__ x2) {
    size_t tid = (size_t)blockIdx.x * blockDim.x + threadIdx.x;
    if (tid >= (size_t)BATCH * SEQ * DMODEL) return;
    int d = (int)(tid % DMODEL);
    int s = (int)((tid / DMODEL) % SEQ);
    int b = (int)(tid / ((size_t)SEQ * DMODEL));
    int tok = x[b * SEQ + s];
    float v = emb[tok * DMODEL + d] + pos[s * DMODEL + d];
    x1[tid] = v;
    x2[tid] = v;
}

// ---------------------------------------------------------------- layernorm -> f16
__global__ void k_ln(const float* __restrict__ X, const float* __restrict__ g,
                     const float* __restrict__ bta, half_t* __restrict__ Y) {
    __shared__ float red[DMODEL];
    int row = blockIdx.x;
    int t = threadIdx.x;
    float v = X[(size_t)row * DMODEL + t];
    red[t] = v;
    __syncthreads();
    for (int off = DMODEL / 2; off > 0; off >>= 1) {
        if (t < off) red[t] += red[t + off];
        __syncthreads();
    }
    float mean = red[0] * (1.0f / DMODEL);
    __syncthreads();
    float dv = v - mean;
    red[t] = dv * dv;
    __syncthreads();
    for (int off = DMODEL / 2; off > 0; off >>= 1) {
        if (t < off) red[t] += red[t + off];
        __syncthreads();
    }
    float var = red[0] * (1.0f / DMODEL);
    float y = dv * rsqrtf(var + 1e-5f) * g[t] + bta[t];
    Y[(size_t)row * DMODEL + t] = (half_t)y;
}

// ---------------------------------------------------------------- W[K][N] f32 -> Wt[N][K] f16
__global__ void k_convT(const float* __restrict__ W, half_t* __restrict__ Wt, int K, int N) {
    size_t tid = (size_t)blockIdx.x * blockDim.x + threadIdx.x;
    if (tid >= (size_t)K * N) return;
    int n = (int)(tid % N);
    int k = (int)(tid / N);
    Wt[(size_t)n * K + k] = (half_t)W[tid];
}

// ---------------------------------------------------------------- WMMA GEMM: C = A[M,K] * Bt[N,K]^T
// block = 256 threads = 8 waves; tile M=128 (wave-per-16-rows), N=64.
// B panel (64 x KC f16, 16 KB) is double-buffered in LDS via
// global_load_async_to_lds_b128; the next chunk's DMA overlaps the current
// chunk's WMMAs. A streams from global.
// Grids guarantee M % 128 == 0, N % 64 == 0, K % KC == 0.
__global__ __launch_bounds__(256) void k_gemm(const half_t* __restrict__ A,
                                              const half_t* __restrict__ Bt,
                                              int M, int N, int K,
                                              float* outF, half_t* outH,
                                              const float* bias, int flags) {
    __shared__ half_t bs[2][64 * KC];
    int wave = threadIdx.x >> 5;
    int lane = threadIdx.x & 31;
    int m0 = (blockIdx.y * 8 + wave) * 16;
    int n0 = blockIdx.x * 64;
    int lmod = lane & 15;
    int ah = (lane & 16) ? 16 : 0;   // A-fragment K offset
    int bk = (lane & 16) ? 16 : 0;   // B-fragment K offset
    v8f acc[4] = {};
    const half_t* arow = A + (size_t)(m0 + lmod) * K;

    auto stage = [&](int kb0, int buf) {
#pragma unroll
        for (int it = 0; it < 4; ++it) {
            int sgid = (int)threadIdx.x + 256 * it;      // 1024 segments of 8 halves
            int nn = sgid >> 4;
            int sg = sgid & 15;
            const half_t* gp = Bt + (size_t)(n0 + nn) * K + kb0 + sg * 8;
            unsigned ldsa = (unsigned)(uintptr_t)(&bs[buf][nn * KC + sg * 8]);
            unsigned long long ga = (unsigned long long)gp;
            asm volatile("global_load_async_to_lds_b128 %0, %1, off"
                         :: "v"(ldsa), "v"(ga) : "memory");
        }
    };

    int nchunks = K / KC;
    stage(0, 0);
    for (int c = 0; c < nchunks; ++c) {
        // current buffer fully written (this wave) + visible to all (barrier);
        // barrier also guarantees everyone is done reading the other buffer.
        asm volatile("s_wait_asynccnt 0x0" ::: "memory");
        __syncthreads();
        if (c + 1 < nchunks) {
            stage((c + 1) * KC, (c + 1) & 1);            // DMA overlaps compute below
            __builtin_prefetch(arow + (c + 1) * KC, 0, 0);
        }
        const half_t* bpan = &bs[c & 1][0];
        int kb0 = c * KC;
        for (int kb = 0; kb < KC; kb += 32) {
            v16h a;
#pragma unroll
            for (int e = 0; e < 16; ++e) a[e] = arow[kb0 + kb + ah + e];
            v16h bb[4];
#pragma unroll
            for (int tt = 0; tt < 4; ++tt) {
                const half_t* brow = bpan + (tt * 16 + lmod) * KC + kb + bk;
#pragma unroll
                for (int e = 0; e < 16; ++e) bb[tt][e] = brow[e];
            }
#pragma unroll
            for (int tt = 0; tt < 4; ++tt)
                acc[tt] = __builtin_amdgcn_wmma_f32_16x16x32_f16(false, a, false, bb[tt],
                                                                 (short)0, acc[tt], false, false);
        }
    }
#pragma unroll
    for (int tt = 0; tt < 4; ++tt) {
#pragma unroll
        for (int r = 0; r < 8; ++r) {
            int m = m0 + r + ((lane & 16) ? 8 : 0);
            int n = n0 + tt * 16 + lmod;
            float v = acc[tt][r];
            if (bias) v += bias[n];
            if (flags & FLAG_GELU) v = gelu_f(v);
            size_t o = (size_t)m * N + n;
            if (outF) {
                if (flags & FLAG_ACCUM) outF[o] += v; else outF[o] = v;
            }
            if (outH) outH[o] = (half_t)v;
        }
    }
}

// ---------------------------------------------------------------- LSH hashing -> packed sort keys
__global__ void k_hash(const float* __restrict__ qk, const float* __restrict__ rot,
                       unsigned* __restrict__ keys) {
    int tid = blockIdx.x * blockDim.x + threadIdx.x;
    if (tid >= BATCH * NHEAD * NHASH * SEQ) return;
    int s = tid & (SEQ - 1);
    int n = (tid >> 12) & (NHASH - 1);
    int h = (tid >> 13) & (NHEAD - 1);
    int b = tid >> 16;
    const float* q = qk + ((size_t)(b * SEQ + s)) * DMODEL + h * DH;
    float qv[DH];
#pragma unroll
    for (int d = 0; d < DH; ++d) qv[d] = q[d];
    const float* R = rot + (size_t)n * DH * (NBUCK / 2);
    float r[NBUCK / 2];
    for (int j = 0; j < NBUCK / 2; ++j) {
        float acc = 0.0f;
#pragma unroll
        for (int d = 0; d < DH; ++d) acc += qv[d] * R[d * (NBUCK / 2) + j];
        r[j] = acc;
    }
    float best = -3.402823466e38f;
    int bi = 0;
    for (int j = 0; j < NBUCK; ++j) {
        float val = (j < NBUCK / 2) ? r[j] : -r[j - NBUCK / 2];
        if (val > best) { best = val; bi = j; }
    }
    int bucket = bi + n * NBUCK;
    int i = n * SEQ + s;
    unsigned key = ((unsigned)(bucket * SEQ + s) << 13) | (unsigned)i;
    keys[((size_t)(b * NHEAD + h)) * SEQ2 + i] = key;
}

// ---------------------------------------------------------------- bitonic sort (per b,h) in LDS
__global__ __launch_bounds__(1024) void k_sort(const unsigned* __restrict__ keys,
                                               unsigned* __restrict__ st,
                                               unsigned* __restrict__ undo) {
    __shared__ unsigned lds[SEQ2];
    int bh = blockIdx.x;
    const unsigned* kp = keys + (size_t)bh * SEQ2;
    for (int i = threadIdx.x; i < SEQ2; i += 1024) lds[i] = kp[i];
    for (int k = 2; k <= SEQ2; k <<= 1) {
        for (int j = k >> 1; j > 0; j >>= 1) {
            __syncthreads();
            for (int i = threadIdx.x; i < SEQ2; i += 1024) {
                int p = i ^ j;
                if (p > i) {
                    bool asc = ((i & k) == 0);
                    unsigned a = lds[i], b = lds[p];
                    if ((a > b) == asc) { lds[i] = b; lds[p] = a; }
                }
            }
        }
    }
    __syncthreads();
    for (int j = threadIdx.x; j < SEQ2; j += 1024) {
        unsigned io = lds[j] & 8191u;
        st[(size_t)bh * SEQ2 + j] = io & (SEQ - 1);
        undo[(size_t)bh * SEQ2 + io] = (unsigned)j;
    }
}

// ---------------------------------------------------------------- gather sorted q / normalized k / v^T
__global__ void k_gather(const float* __restrict__ qk, const half_t* __restrict__ vh,
                         const unsigned* __restrict__ st, half_t* __restrict__ sq,
                         half_t* __restrict__ skn, half_t* __restrict__ svt) {
    int tid = blockIdx.x * blockDim.x + threadIdx.x;
    if (tid >= BATCH * NHEAD * SEQ2) return;
    int j = tid & (SEQ2 - 1);
    int bh = tid >> 13;
    int h = bh & (NHEAD - 1);
    int b = bh >> 3;
    unsigned pos = st[(size_t)bh * SEQ2 + j];
    const float* qrow = qk + ((size_t)(b * SEQ + pos)) * DMODEL + h * DH;
    float q[DH];
    float nrm = 0.0f;
#pragma unroll
    for (int d = 0; d < DH; ++d) { q[d] = qrow[d]; nrm += q[d] * q[d]; }
    nrm = fmaxf(sqrtf(nrm), 1e-6f);
    float inv = 1.0f / nrm;
    size_t base = ((size_t)bh * SEQ2 + j) * DH;
#pragma unroll
    for (int d = 0; d < DH; ++d) {
        sq[base + d] = (half_t)q[d];
        skn[base + d] = (half_t)(q[d] * inv);
    }
    const half_t* vrow = vh + ((size_t)(b * SEQ + pos)) * DMODEL + h * DH;
#pragma unroll
    for (int d = 0; d < DH; ++d)
        svt[((size_t)bh * DH + d) * SEQ2 + j] = vrow[d];
}

// ---------------------------------------------------------------- bucketed attention (WMMA)
// block = one (b,h,chunk); 128 threads = 4 waves, each wave = one 16-row M tile.
__global__ __launch_bounds__(128) void k_attn(const half_t* __restrict__ sq,
                                              const half_t* __restrict__ skn,
                                              const half_t* __restrict__ svt,
                                              const unsigned* __restrict__ st,
                                              half_t* __restrict__ soh,
                                              float* __restrict__ slog) {
    __shared__ float dots[BUCKSZ][2 * BUCKSZ];
    __shared__ half_t pbuf[BUCKSZ][2 * BUCKSZ];
    __shared__ unsigned stq[BUCKSZ];
    __shared__ unsigned stk[2 * BUCKSZ];
    int c = blockIdx.x & (NCH - 1);
    int bh = blockIdx.x >> 7;
    int cbase = c * BUCKSZ;
    int pbase = ((c + NCH - 1) & (NCH - 1)) * BUCKSZ;
    const unsigned* stbh = st + (size_t)bh * SEQ2;
    int t = threadIdx.x;
    stk[t] = stbh[(t < BUCKSZ) ? (cbase + t) : (pbase + (t - BUCKSZ))];
    if (t < BUCKSZ) stq[t] = stk[t];
    __syncthreads();

    int wave = t >> 5, lane = t & 31;
    int m0 = wave * 16;
    int lmod = lane & 15;
    int ah = (lane & 16) ? 16 : 0;
    int bk = (lane & 16) ? 16 : 0;
    const half_t* sqb = sq + (size_t)bh * SEQ2 * DH;
    const half_t* sknb = skn + (size_t)bh * SEQ2 * DH;
    const half_t* svb = svt + (size_t)bh * DH * SEQ2;

    // ---- dots = bq @ bkn^T, scaled + masked -> LDS
    v16h a;
    {
        const half_t* ap = sqb + (size_t)(cbase + m0 + lmod) * DH + ah;
#pragma unroll
        for (int e = 0; e < 16; ++e) a[e] = ap[e];
    }
    const float scale = 0.17677669529663687f; // 32^-0.5
    for (int tt = 0; tt < 8; ++tt) {
        int kk = tt * 16 + lmod;
        int jr = (kk < BUCKSZ) ? (cbase + kk) : (pbase + kk - BUCKSZ);
        v16h bb;
        const half_t* bp = sknb + (size_t)jr * DH + bk;
#pragma unroll
        for (int e = 0; e < 16; ++e) bb[e] = bp[e];
        v8f acc = {};
        acc = __builtin_amdgcn_wmma_f32_16x16x32_f16(false, a, false, bb, (short)0, acc, false, false);
#pragma unroll
        for (int r = 0; r < 8; ++r) {
            int mi = m0 + r + ((lane & 16) ? 8 : 0);
            int ni = tt * 16 + lmod;
            float v = acc[r] * scale;
            unsigned qp = stq[mi], kp = stk[ni];
            if (qp < kp) v = -1e9f;
            else if (qp == kp) v = -1e5f;
            dots[mi][ni] = v;
        }
    }
    __syncthreads();

    // ---- softmax rows (logsumexp), write P f16 + slog
    if (t < BUCKSZ) {
        float mx = -3.402823466e38f;
        for (int n = 0; n < 2 * BUCKSZ; ++n) mx = fmaxf(mx, dots[t][n]);
        float sum = 0.0f;
        for (int n = 0; n < 2 * BUCKSZ; ++n) sum += expf(dots[t][n] - mx);
        float lse = mx + logf(sum);
        for (int n = 0; n < 2 * BUCKSZ; ++n) pbuf[t][n] = (half_t)expf(dots[t][n] - lse);
        slog[(size_t)bh * SEQ2 + cbase + t] = lse;
    }
    __syncthreads();

    // ---- bo = P @ bv  (K = 128 in 4 steps, N = 32 in 2 tiles)
    v8f oacc[2] = {};
    for (int ks = 0; ks < 4; ++ks) {
        v16h pa;
        int row = m0 + lmod;
#pragma unroll
        for (int e = 0; e < 16; ++e) pa[e] = pbuf[row][ks * 32 + ah + e];
#pragma unroll
        for (int nt = 0; nt < 2; ++nt) {
            int ndh = nt * 16 + lmod;
            v16h vb;
#pragma unroll
            for (int e = 0; e < 16; ++e) {
                int key = ks * 32 + bk + e;
                int jr = (key < BUCKSZ) ? (cbase + key) : (pbase + key - BUCKSZ);
                vb[e] = svb[(size_t)ndh * SEQ2 + jr];
            }
            oacc[nt] = __builtin_amdgcn_wmma_f32_16x16x32_f16(false, pa, false, vb,
                                                              (short)0, oacc[nt], false, false);
        }
    }
#pragma unroll
    for (int nt = 0; nt < 2; ++nt) {
#pragma unroll
        for (int r = 0; r < 8; ++r) {
            int mi = m0 + r + ((lane & 16) ? 8 : 0);
            int ndh = nt * 16 + lmod;
            soh[((size_t)bh * SEQ2 + cbase + mi) * DH + ndh] = (half_t)oacc[nt][r];
        }
    }
}

// ---------------------------------------------------------------- unsort + combine hash rounds
__global__ void k_combine(const half_t* __restrict__ soh, const float* __restrict__ slog,
                          const unsigned* __restrict__ undo, half_t* __restrict__ att) {
    int tid = blockIdx.x * blockDim.x + threadIdx.x;
    if (tid >= BATCH * SEQ * NHEAD) return;
    int h = tid & (NHEAD - 1);
    int i = (tid >> 3) & (SEQ - 1);
    int b = tid >> 15;
    int bh = b * NHEAD + h;
    const unsigned* ub = undo + (size_t)bh * SEQ2;
    unsigned j0 = ub[i], j1 = ub[SEQ + i];
    float l0 = slog[(size_t)bh * SEQ2 + j0];
    float l1 = slog[(size_t)bh * SEQ2 + j1];
    float mx = fmaxf(l0, l1);
    float w0 = expf(l0 - mx), w1 = expf(l1 - mx);
    float inv = 1.0f / (w0 + w1);
    const half_t* o0 = soh + ((size_t)bh * SEQ2 + j0) * DH;
    const half_t* o1 = soh + ((size_t)bh * SEQ2 + j1) * DH;
    half_t* out = att + ((size_t)(b * SEQ + i)) * DMODEL + h * DH;
#pragma unroll
    for (int d = 0; d < DH; ++d)
        out[d] = (half_t)((w0 * (float)o0[d] + w1 * (float)o1[d]) * inv);
}

// ---------------------------------------------------------------- classifier head
__global__ void k_final(const float* __restrict__ x1, const float* __restrict__ x2,
                        const float* __restrict__ Wf, const float* __restrict__ bf,
                        float* __restrict__ out) {
    __shared__ float red[256];
    int c = blockIdx.x, b = blockIdx.y, t = threadIdx.x;
    const size_t NF = (size_t)SEQ * DMODEL;
    const float* p1 = x1 + (size_t)b * NF;
    const float* p2 = x2 + (size_t)b * NF;
    float acc = 0.0f;
    for (size_t i = t; i < NF; i += 256)
        acc += 0.5f * (p1[i] + p2[i]) * Wf[i * NCLASS + c];
    red[t] = acc;
    __syncthreads();
    for (int off = 128; off > 0; off >>= 1) {
        if (t < off) red[t] += red[t + off];
        __syncthreads();
    }
    if (t == 0) out[b * NCLASS + c] = red[0] + bf[c];
}

// ================================================================ host
extern "C" void kernel_launch(void* const* d_in, const int* in_sizes, int n_in,
                              void* d_out, int out_size, void* d_ws, size_t ws_size,
                              hipStream_t stream) {
    const int*   x      = (const int*)d_in[0];
    const float* emb    = (const float*)d_in[1];
    const float* posenc = (const float*)d_in[2];
    const float* ln1_g  = (const float*)d_in[3];
    const float* ln1_b  = (const float*)d_in[4];
    const float* Wqk    = (const float*)d_in[5];
    const float* Wv     = (const float*)d_in[6];
    const float* Wo     = (const float*)d_in[7];
    const float* ln2_g  = (const float*)d_in[8];
    const float* ln2_b  = (const float*)d_in[9];
    const float* W1     = (const float*)d_in[10];
    const float* b1     = (const float*)d_in[11];
    const float* W2     = (const float*)d_in[12];
    const float* b2     = (const float*)d_in[13];
    const float* rot    = (const float*)d_in[14];
    const float* Wf     = (const float*)d_in[15];
    const float* bf     = (const float*)d_in[16];
    float* out = (float*)d_out;

    char* ws = (char*)d_ws;
    size_t off = 0;
    auto alloc = [&](size_t bytes) -> void* {
        void* p = (void*)(ws + off);
        off = (off + bytes + 255) & ~(size_t)255;
        return p;
    };
    const size_t actF = (size_t)MROWS * DMODEL * sizeof(float);     // 32 MB
    const size_t actH = (size_t)MROWS * DMODEL * sizeof(half_t);    // 16 MB
    const size_t srtH = (size_t)BATCH * NHEAD * SEQ2 * DH * sizeof(half_t); // 32 MB
    const size_t idxU = (size_t)BATCH * NHEAD * SEQ2 * sizeof(unsigned);    // 2 MB

    float*    x1    = (float*)alloc(actF);
    float*    x2    = (float*)alloc(actF);
    float*    qk    = (float*)alloc(actF);
    half_t*   vh    = (half_t*)alloc(actH);
    half_t*   xnh   = (half_t*)alloc(actH);
    half_t*   atth  = (half_t*)alloc(actH);
    unsigned* keys  = (unsigned*)alloc(idxU);
    unsigned* st    = (unsigned*)alloc(idxU);
    unsigned* undo  = (unsigned*)alloc(idxU);
    half_t*   soh   = (half_t*)alloc(srtH);
    float*    slogv = (float*)alloc(idxU);
    half_t*   wqkT  = (half_t*)alloc((size_t)DMODEL * DMODEL * sizeof(half_t));
    half_t*   wvT   = (half_t*)alloc((size_t)DMODEL * DMODEL * sizeof(half_t));
    half_t*   woT   = (half_t*)alloc((size_t)DMODEL * DMODEL * sizeof(half_t));
    half_t*   w1T   = (half_t*)alloc((size_t)DMODEL * DFF * sizeof(half_t));
    half_t*   w2T   = (half_t*)alloc((size_t)DFF * DMODEL * sizeof(half_t));
    // union region: {sq, skn, svt} (attention phase) aliases gelu buffer (FFN phase)
    char* uni = (char*)alloc(3 * srtH);
    half_t* sqh  = (half_t*)uni;
    half_t* sknh = (half_t*)(uni + srtH);
    half_t* svth = (half_t*)(uni + 2 * srtH);
    half_t* gh   = (half_t*)uni;   // MROWS * DFF * 2 = 64 MB <= 96 MB
    (void)ws_size; (void)in_sizes; (void)n_in; (void)out_size;

    // embedding + stream init
    {
        size_t total = (size_t)BATCH * SEQ * DMODEL;
        k_embed<<<dim3((unsigned)((total + 255) / 256)), dim3(256), 0, stream>>>(x, emb, posenc, x1, x2);
    }

    for (int l = 0; l < NLAYERS; ++l) {
        const float* wqk_l = Wqk + (size_t)l * DMODEL * DMODEL;
        const float* wv_l  = Wv  + (size_t)l * DMODEL * DMODEL;
        const float* wo_l  = Wo  + (size_t)l * DMODEL * DMODEL;
        const float* w1_l  = W1  + (size_t)l * DMODEL * DFF;
        const float* w2_l  = W2  + (size_t)l * DFF * DMODEL;
        const float* rot_l = rot + (size_t)l * NHASH * DH * (NBUCK / 2);

        k_convT<<<dim3(256), dim3(256), 0, stream>>>(wqk_l, wqkT, DMODEL, DMODEL);
        k_convT<<<dim3(256), dim3(256), 0, stream>>>(wv_l, wvT, DMODEL, DMODEL);
        k_convT<<<dim3(256), dim3(256), 0, stream>>>(wo_l, woT, DMODEL, DMODEL);
        k_convT<<<dim3(1024), dim3(256), 0, stream>>>(w1_l, w1T, DMODEL, DFF);
        k_convT<<<dim3(1024), dim3(256), 0, stream>>>(w2_l, w2T, DFF, DMODEL);

        // ln1(x2) -> xnh
        k_ln<<<dim3(MROWS), dim3(DMODEL), 0, stream>>>(x2, ln1_g + l * DMODEL, ln1_b + l * DMODEL, xnh);
        // qk = xnh @ Wqk (f32); v = xnh @ Wv (f16)
        k_gemm<<<dim3(DMODEL / 64, MROWS / 128), dim3(256), 0, stream>>>(
            xnh, wqkT, MROWS, DMODEL, DMODEL, qk, nullptr, nullptr, 0);
        k_gemm<<<dim3(DMODEL / 64, MROWS / 128), dim3(256), 0, stream>>>(
            xnh, wvT, MROWS, DMODEL, DMODEL, nullptr, vh, nullptr, 0);
        // LSH hash -> keys; bitonic sort -> st/undo; gather sorted tensors
        k_hash<<<dim3(BATCH * NHEAD * NHASH * SEQ / 256), dim3(256), 0, stream>>>(qk, rot_l, keys);
        k_sort<<<dim3(BATCH * NHEAD), dim3(1024), 0, stream>>>(keys, st, undo);
        k_gather<<<dim3(BATCH * NHEAD * SEQ2 / 256), dim3(256), 0, stream>>>(qk, vh, st, sqh, sknh, svth);
        // bucketed attention
        k_attn<<<dim3(BATCH * NHEAD * NCH), dim3(128), 0, stream>>>(sqh, sknh, svth, st, soh, slogv);
        // unsort + combine hash rounds -> atth ; x1 += atth @ Wo
        k_combine<<<dim3(BATCH * SEQ * NHEAD / 256), dim3(256), 0, stream>>>(soh, slogv, undo, atth);
        k_gemm<<<dim3(DMODEL / 64, MROWS / 128), dim3(256), 0, stream>>>(
            atth, woT, MROWS, DMODEL, DMODEL, x1, nullptr, nullptr, FLAG_ACCUM);

        // FFN: x2 += gelu(ln2(x1) @ W1 + b1) @ W2 + b2
        k_ln<<<dim3(MROWS), dim3(DMODEL), 0, stream>>>(x1, ln2_g + l * DMODEL, ln2_b + l * DMODEL, xnh);
        k_gemm<<<dim3(DFF / 64, MROWS / 128), dim3(256), 0, stream>>>(
            xnh, w1T, MROWS, DFF, DMODEL, nullptr, gh, b1 + l * DFF, FLAG_GELU);
        k_gemm<<<dim3(DMODEL / 64, MROWS / 128), dim3(256), 0, stream>>>(
            gh, w2T, MROWS, DMODEL, DFF, x2, nullptr, b2 + l * DMODEL, FLAG_ACCUM);
    }

    k_final<<<dim3(NCLASS, BATCH), dim3(256), 0, stream>>>(x1, x2, Wf, bf, out);
}
